// EdgeAwareGAT_16870631539205
// MI455X (gfx1250) — matrix operand order repaired and checked by reading, not compiled, not run on hardware
//
#include <hip/hip_runtime.h>
#include <hip/hip_bf16.h>
#include <cstdint>
#include <math.h>

#define NN 10000
#define NE 160000
#define NG 64
#define D_IN 771
#define HIDDEN 256
#define NHD 4

typedef __attribute__((ext_vector_type(16))) __bf16 v16bf;
typedef __attribute__((ext_vector_type(8)))  float  v8f;

// ---------- helpers ----------
static __device__ __forceinline__ unsigned short f2bf(float f) {
  unsigned int u = __float_as_uint(f);
  u += 0x7FFFu + ((u >> 16) & 1u);           // round-to-nearest-even
  return (unsigned short)(u >> 16);
}
// monotone float<->uint encoding so atomicMax(u32) == float max
static __device__ __forceinline__ unsigned fkey(float f) {
  unsigned u = __float_as_uint(f);
  return (u & 0x80000000u) ? ~u : (u | 0x80000000u);
}
static __device__ __forceinline__ float fdec(unsigned k) {
  unsigned u = (k & 0x80000000u) ? (k ^ 0x80000000u) : ~k;
  return __uint_as_float(u);
}
static __device__ __forceinline__ float elu1(float v) {
  return v > 0.f ? v : (__expf(v) - 1.f);
}

// ---------- fp32 -> bf16 with K padding ----------
__global__ void convert_pad_bf16(const float* __restrict__ src,
                                 unsigned short* __restrict__ dst,
                                 int rows, int korig, int kpad) {
  size_t total = (size_t)rows * kpad;
  size_t stride = (size_t)gridDim.x * blockDim.x;
  for (size_t o = (size_t)blockIdx.x * blockDim.x + threadIdx.x; o < total; o += stride) {
    size_t r = o / kpad;
    int k = (int)(o - r * (size_t)kpad);
    dst[o] = (k < korig) ? f2bf(src[r * (size_t)korig + k]) : (unsigned short)0;
  }
}

// ---------- pack W[K,M] fp32 into per-lane WMMA B-fragment bf16 layout ----------
// For K-tile kt, col-tile ct, lane l: 16 contiguous halves = column (ct*16 + l%16),
// K = kt*32 + (l>=16 ? 16 : 0) + j, j=0..15  (contiguous k order, same as the A fragments)
__global__ void pack_w_bf16(const float* __restrict__ W,
                            unsigned short* __restrict__ Bp,
                            int korig, int kpad, int M) {
  size_t total = (size_t)kpad * M;
  size_t stride = (size_t)gridDim.x * blockDim.x;
  int nCT = M >> 4;
  for (size_t o = (size_t)blockIdx.x * blockDim.x + threadIdx.x; o < total; o += stride) {
    int j = (int)(o & 15);
    size_t t = o >> 4;
    int lane = (int)(t & 31);
    t >>= 5;
    int ct = (int)(t % nCT);
    int kt = (int)(t / nCT);
    int col = (ct << 4) + (lane & 15);
    int k = (kt << 5) + ((lane >> 4) << 4) + j;
    float v = (k < korig) ? W[(size_t)k * M + col] : 0.f;
    Bp[o] = f2bf(v);
  }
}

// ---------- bf16 WMMA GEMM with async-LDS A-tile staging ----------
// Block = 8 waves, one 16-row tile per block, 2 col-tiles per wave (no idle waves:
// M=256 -> grid.x=1, M=1024 -> grid.x=4). A tile (16x32 bf16 = 1KB) is staged into
// LDS once per K-step by waves 0-1 via global_load_async_to_lds_b128 (double
// buffered, ASYNCcnt), then all 8 waves read fragments from LDS (ds_load_b128).
__global__ __launch_bounds__(256) void gemm_bf16_wmma(
    const unsigned short* __restrict__ A,   // [rows, kpad] bf16 row-major, rows%16==0
    const unsigned short* __restrict__ Bp,  // packed fragments
    float* __restrict__ C, int kpad, int M) {
  __shared__ __align__(16) unsigned short smA[2][16][32];
  const int t = threadIdx.x;
  const int lane = t & 31;
  const int wave = t >> 5;
  const int nCT = M >> 4;
  const int rt = blockIdx.y;
  const int ct0 = (blockIdx.x * 8 + wave) * 2;
  const int hi = (lane >> 4) & 1;
  const int nKT = kpad >> 5;

  // async copy mapping for t<64: 64 lanes x 16B = 1KB tile
  const int arow = t >> 2, aseg = t & 3;
  const unsigned short* asrc =
      A + (size_t)((rt << 4) + arow) * kpad + aseg * 8;
  const unsigned ldsb0 = (unsigned)(uintptr_t)(&smA[0][arow][aseg * 8]);

  if (t < 64) {  // wave-uniform branch (waves 0 and 1)
    unsigned long long g = (unsigned long long)(uintptr_t)asrc;
    asm volatile("global_load_async_to_lds_b128 %0, %1, off"
                 :: "v"(ldsb0), "v"(g) : "memory");
  }

  v8f acc0 = {0.f, 0.f, 0.f, 0.f, 0.f, 0.f, 0.f, 0.f};
  v8f acc1 = {0.f, 0.f, 0.f, 0.f, 0.f, 0.f, 0.f, 0.f};
  union Frag { uint4 q[2]; v16bf v; };

  for (int kt = 0; kt < nKT; ++kt) {
    const int buf = kt & 1;
    if (t < 64) {
      asm volatile("s_wait_asynccnt 0x0" ::: "memory");
    }
    __syncthreads();  // staged A tile visible to all 8 waves
    if (t < 64 && (kt + 1) < nKT) {  // prefetch next A tile into other buffer
      unsigned long long g =
          (unsigned long long)(uintptr_t)(asrc + ((size_t)(kt + 1) << 5));
      unsigned l = ldsb0 + (((kt + 1) & 1) ? 1024u : 0u);
      asm volatile("global_load_async_to_lds_b128 %0, %1, off"
                   :: "v"(l), "v"(g) : "memory");
    }

    Frag fa, fb0, fb1;
    const unsigned short* ap = &smA[buf][lane & 15][hi * 16];
    fa.q[0] = *(const uint4*)(ap);         // K = hi*16 .. hi*16+7
    fa.q[1] = *(const uint4*)(ap + 8);     // K = hi*16+8 .. hi*16+15

    const unsigned short* bp0 =
        Bp + ((((size_t)kt * nCT + ct0) << 5) + lane) * 16;
    fb0.q[0] = *(const uint4*)(bp0);
    fb0.q[1] = *(const uint4*)(bp0 + 8);
    fb1.q[0] = *(const uint4*)(bp0 + 512);      // next col tile: +32*16 halves
    fb1.q[1] = *(const uint4*)(bp0 + 512 + 8);
    __builtin_prefetch(bp0 + (size_t)nCT * 512, 0, 0);  // next kt's B fragments

    acc0 = __builtin_amdgcn_wmma_f32_16x16x32_bf16(false, fa.v, false, fb0.v,
                                                   (short)0, acc0, false, false);
    acc1 = __builtin_amdgcn_wmma_f32_16x16x32_bf16(false, fa.v, false, fb1.v,
                                                   (short)0, acc1, false, false);
    __syncthreads();  // all reads of smA[buf] done before it is refilled
  }

  float* crow = C + (size_t)((rt << 4) + hi * 8) * M + (ct0 << 4) + (lane & 15);
#pragma unroll
  for (int r = 0; r < 8; ++r) {
    crow[(size_t)r * M] = acc0[r];
    crow[(size_t)r * M + 16] = acc1[r];
  }
}

// ---------- fold edge-attn vector: ve[d*4+h] = sum_f We[d, h*F+f]*ae[h,f] ----------
__global__ void fold_edge_att(const float* __restrict__ We, const float* __restrict__ ae,
                              float* __restrict__ ve, int M, int F) {
  int t = threadIdx.x;
  if (t >= 16) return;
  int d = t >> 2, h = t & 3;
  float s = 0.f;
  for (int f = 0; f < F; ++f) s += We[(size_t)d * M + h * F + f] * ae[h * F + f];
  ve[d * 4 + h] = s;
}

// ---------- per-node attention scores sl/sr [N,H] ----------
__global__ void node_scores(const float* __restrict__ hlin,
                            const float* __restrict__ a_s, const float* __restrict__ a_d,
                            float* __restrict__ sl, float* __restrict__ sr, int M, int F) {
  int idx = blockIdx.x * blockDim.x + threadIdx.x;
  if (idx >= NN * NHD) return;
  int n = idx >> 2, h = idx & 3;
  const float* hp = hlin + (size_t)n * M + h * F;
  const float* s0 = a_s + h * F;
  const float* d0 = a_d + h * F;
  float a = 0.f, b = 0.f;
  for (int f = 0; f < F; ++f) { float hv = hp[f]; a += hv * s0[f]; b += hv * d0[f]; }
  sl[idx] = a;
  sr[idx] = b;
}

// ---------- edge logits + segment max (ordered-uint atomicMax) ----------
__global__ void edge_logits(const int* __restrict__ ei, const float* __restrict__ ea,
                            const float* __restrict__ sl, const float* __restrict__ sr,
                            const float* __restrict__ ve,
                            float* __restrict__ al, unsigned* __restrict__ mkey) {
  int e = blockIdx.x * blockDim.x + threadIdx.x;
  if (e >= NE) return;
  int s = ei[e], d = ei[NE + e];
  float e0 = ea[e * 4 + 0], e1 = ea[e * 4 + 1], e2 = ea[e * 4 + 2], e3 = ea[e * 4 + 3];
#pragma unroll
  for (int h = 0; h < 4; ++h) {
    float v = sl[s * 4 + h] + sr[d * 4 + h] +
              e0 * ve[0 + h] + e1 * ve[4 + h] + e2 * ve[8 + h] + e3 * ve[12 + h];
    v = v > 0.f ? v : 0.2f * v;           // leaky_relu 0.2
    al[e * 4 + h] = v;
    atomicMax(&mkey[d * 4 + h], fkey(v));
  }
}

// ---------- exp(al - m[dst]) + segment sum of denominators ----------
__global__ void edge_exp(const int* __restrict__ ei, const float* __restrict__ al,
                         const unsigned* __restrict__ mkey,
                         float* __restrict__ exv, float* __restrict__ den) {
  int e = blockIdx.x * blockDim.x + threadIdx.x;
  if (e >= NE) return;
  int d = ei[NE + e];
#pragma unroll
  for (int h = 0; h < 4; ++h) {
    float x = __expf(al[e * 4 + h] - fdec(mkey[d * 4 + h]));
    exv[e * 4 + h] = x;
    atomicAdd(&den[d * 4 + h], x);
  }
}

// ---------- weighted scatter, float4 per thread: agg[dst,i..i+3] += h[src]*alpha ----------
__global__ void aggregate(const int* __restrict__ ei, const float* __restrict__ hlin,
                          const float* __restrict__ exv, const float* __restrict__ den,
                          float* __restrict__ agg, int M, int shiftF) {
  const int mq = M >> 2;
  size_t total = (size_t)NE * mq;
  size_t stride = (size_t)gridDim.x * blockDim.x;
  for (size_t o = (size_t)blockIdx.x * blockDim.x + threadIdx.x; o < total; o += stride) {
    size_t e = o / (size_t)mq;
    int i = ((int)(o - e * (size_t)mq)) << 2;
    int h = i >> shiftF;                  // 4-group never crosses a head (F=64/256)
    int s = ei[e], d = ei[NE + e];
    float w = exv[e * 4 + h] / (den[d * 4 + h] + 1e-16f);
    const float4 hv = *(const float4*)&hlin[(size_t)s * M + i];
    float* dp = &agg[(size_t)d * M + i];
    atomicAdd(dp + 0, hv.x * w);
    atomicAdd(dp + 1, hv.y * w);
    atomicAdd(dp + 2, hv.z * w);
    atomicAdd(dp + 3, hv.w * w);
  }
}

// ---------- epilogues ----------
__global__ void epilogue_concat(const float* __restrict__ agg, const float* __restrict__ b,
                                float* __restrict__ act) {
  int idx = blockIdx.x * blockDim.x + threadIdx.x;
  if (idx >= NN * HIDDEN) return;
  act[idx] = elu1(agg[idx] + b[idx & 255]);
}
__global__ void epilogue_mean(const float* __restrict__ agg, const float* __restrict__ b,
                              float* __restrict__ act) {
  int idx = blockIdx.x * blockDim.x + threadIdx.x;
  if (idx >= NN * HIDDEN) return;
  int n = idx >> 8, f = idx & 255;
  const float* a = agg + (size_t)n * 1024 + f;
  float v = 0.25f * (a[0] + a[256] + a[512] + a[768]) + b[f];
  act[idx] = elu1(v);
}

// ---------- attention pooling: node score s = tanh(h@pw1+pb1)@pw2+pb2 ----------
__global__ __launch_bounds__(256) void pool_score(
    const float* __restrict__ h3, const float* __restrict__ pw1,
    const float* __restrict__ pb1, const float* __restrict__ pw2,
    const float* __restrict__ pb2, const int* __restrict__ batch,
    float* __restrict__ s, unsigned* __restrict__ mG) {
  __shared__ float hrow[HIDDEN];
  __shared__ float part[128];
  int n = blockIdx.x, t = threadIdx.x;
  hrow[t] = h3[(size_t)n * HIDDEN + t];
  __syncthreads();
  if (t < 128) {
    float a = pb1[t];
    for (int k = 0; k < HIDDEN; ++k) a += hrow[k] * pw1[k * 128 + t];
    part[t] = tanhf(a) * pw2[t];
  }
  __syncthreads();
  for (int st = 64; st > 0; st >>= 1) {
    if (t < st) part[t] += part[t + st];
    __syncthreads();
  }
  if (t == 0) {
    float sv = part[0] + pb2[0];
    s[n] = sv;
    atomicMax(&mG[batch[n]], fkey(sv));
  }
}

__global__ void pool_exp(const float* __restrict__ s, const int* __restrict__ batch,
                         const unsigned* __restrict__ mG,
                         float* __restrict__ exn, float* __restrict__ denG) {
  int n = blockIdx.x * blockDim.x + threadIdx.x;
  if (n >= NN) return;
  int g = batch[n];
  float x = __expf(s[n] - fdec(mG[g]));
  exn[n] = x;
  atomicAdd(&denG[g], x);
}

__global__ void pool_weighted(const float* __restrict__ h3, const int* __restrict__ batch,
                              const float* __restrict__ exn, const float* __restrict__ denG,
                              float* __restrict__ ge) {
  int idx = blockIdx.x * blockDim.x + threadIdx.x;
  if (idx >= NN * HIDDEN) return;
  int n = idx >> 8, f = idx & 255;
  int g = batch[n];
  atomicAdd(&ge[(size_t)g * HIDDEN + f], h3[idx] * (exn[n] / denG[g]));
}

// ---------- classifier head: out[g] = relu(ge@cw1+cb1)@cw2 + cb2 ----------
__global__ __launch_bounds__(256) void classify(
    const float* __restrict__ ge, const float* __restrict__ cw1,
    const float* __restrict__ cb1, const float* __restrict__ cw2,
    const float* __restrict__ cb2, float* __restrict__ out) {
  __shared__ float row[HIDDEN];
  __shared__ float part[128];
  int g = blockIdx.x, t = threadIdx.x;
  row[t] = ge[(size_t)g * HIDDEN + t];
  __syncthreads();
  if (t < 128) {
    float a = cb1[t];
    for (int k = 0; k < HIDDEN; ++k) a += row[k] * cw1[k * 128 + t];
    part[t] = fmaxf(a, 0.f) * cw2[t];
  }
  __syncthreads();
  for (int st = 64; st > 0; st >>= 1) {
    if (t < st) part[t] += part[t + st];
    __syncthreads();
  }
  if (t == 0) out[g] = part[0] + cb2[0];
}

extern "C" void kernel_launch(void* const* d_in, const int* in_sizes, int n_in,
                              void* d_out, int out_size, void* d_ws, size_t ws_size,
                              hipStream_t stream) {
  (void)in_sizes; (void)n_in; (void)out_size; (void)ws_size;
  const float* x   = (const float*)d_in[0];
  const int*   ei  = (const int*)d_in[1];
  const float* ea  = (const float*)d_in[2];
  const int*   bat = (const int*)d_in[3];
  const float* W[3]  = {(const float*)d_in[4],  (const float*)d_in[10], (const float*)d_in[16]};
  const float* As[3] = {(const float*)d_in[5],  (const float*)d_in[11], (const float*)d_in[17]};
  const float* Ad[3] = {(const float*)d_in[6],  (const float*)d_in[12], (const float*)d_in[18]};
  const float* Wev[3]= {(const float*)d_in[7],  (const float*)d_in[13], (const float*)d_in[19]};
  const float* Aev[3]= {(const float*)d_in[8],  (const float*)d_in[14], (const float*)d_in[20]};
  const float* Bv[3] = {(const float*)d_in[9],  (const float*)d_in[15], (const float*)d_in[21]};
  const float* pw1 = (const float*)d_in[22];
  const float* pb1 = (const float*)d_in[23];
  const float* pw2 = (const float*)d_in[24];
  const float* pb2 = (const float*)d_in[25];
  const float* cw1 = (const float*)d_in[26];
  const float* cb1 = (const float*)d_in[27];
  const float* cw2 = (const float*)d_in[28];
  const float* cb2 = (const float*)d_in[29];
  float* out = (float*)d_out;

  uintptr_t p = (uintptr_t)d_ws;
  auto alloc = [&](size_t bytes) -> void* {
    void* r = (void*)p;
    p += (bytes + 255) & ~(size_t)255;
    return r;
  };
  unsigned short* Xb   = (unsigned short*)alloc((size_t)NN * 800 * 2);
  unsigned short* Hb   = (unsigned short*)alloc((size_t)NN * HIDDEN * 2);
  unsigned short* Wp   = (unsigned short*)alloc((size_t)256 * 1024 * 2);
  float*    hlin = (float*)alloc((size_t)NN * 1024 * 4);
  float*    agg  = (float*)alloc((size_t)NN * 1024 * 4);
  float*    act  = (float*)alloc((size_t)NN * HIDDEN * 4);
  float*    alo  = (float*)alloc((size_t)NE * NHD * 4);
  float*    exv  = (float*)alloc((size_t)NE * NHD * 4);
  unsigned* mkey = (unsigned*)alloc((size_t)NN * NHD * 4);
  float*    den  = (float*)alloc((size_t)NN * NHD * 4);
  float*    sl   = (float*)alloc((size_t)NN * NHD * 4);
  float*    sr   = (float*)alloc((size_t)NN * NHD * 4);
  float*    ve   = (float*)alloc(64 * 4);
  float*    sN   = (float*)alloc((size_t)NN * 4);
  float*    exN  = (float*)alloc((size_t)NN * 4);
  unsigned* mG   = (unsigned*)alloc(NG * 4);
  float*    denG = (float*)alloc(NG * 4);
  float*    ge   = (float*)alloc((size_t)NG * HIDDEN * 4);

  const int Ms[3] = {256, 256, 1024};
  const int Fs[3] = {64, 64, 256};
  const int Sh[3] = {6, 6, 8};
  const int Ko[3] = {D_IN, 256, 256};
  const int Kp[3] = {800, 256, 256};

  convert_pad_bf16<<<dim3(1024), dim3(256), 0, stream>>>(x, Xb, NN, D_IN, 800);

  for (int L = 0; L < 3; ++L) {
    const unsigned short* Ain = Xb;
    if (L > 0) {
      convert_pad_bf16<<<dim3(1024), dim3(256), 0, stream>>>(act, Hb, NN, 256, 256);
      Ain = Hb;
    }
    int M = Ms[L], F = Fs[L], kpad = Kp[L];
    pack_w_bf16<<<dim3(512), dim3(256), 0, stream>>>(W[L], Wp, Ko[L], kpad, M);
    dim3 ggrid(M / 256, NN / 16);  // 8 waves x 2 col-tiles = 256 cols per block
    gemm_bf16_wmma<<<ggrid, dim3(256), 0, stream>>>(Ain, Wp, hlin, kpad, M);
    fold_edge_att<<<dim3(1), dim3(16), 0, stream>>>(Wev[L], Aev[L], ve, M, F);
    node_scores<<<dim3((NN * NHD + 255) / 256), dim3(256), 0, stream>>>(
        hlin, As[L], Ad[L], sl, sr, M, F);
    hipMemsetAsync(mkey, 0, (size_t)NN * NHD * 4, stream);  // key 0 == -inf identity
    hipMemsetAsync(den, 0, (size_t)NN * NHD * 4, stream);
    hipMemsetAsync(agg, 0, (size_t)NN * M * 4, stream);
    edge_logits<<<dim3((NE + 255) / 256), dim3(256), 0, stream>>>(
        ei, ea, sl, sr, ve, alo, mkey);
    edge_exp<<<dim3((NE + 255) / 256), dim3(256), 0, stream>>>(ei, alo, mkey, exv, den);
    aggregate<<<dim3(4096), dim3(256), 0, stream>>>(ei, hlin, exv, den, agg, M, Sh[L]);
    if (L < 2)
      epilogue_concat<<<dim3((NN * HIDDEN + 255) / 256), dim3(256), 0, stream>>>(
          agg, Bv[L], act);
    else
      epilogue_mean<<<dim3((NN * HIDDEN + 255) / 256), dim3(256), 0, stream>>>(
          agg, Bv[L], act);
  }

  hipMemsetAsync(mG, 0, NG * 4, stream);
  hipMemsetAsync(denG, 0, NG * 4, stream);
  hipMemsetAsync(ge, 0, (size_t)NG * HIDDEN * 4, stream);
  pool_score<<<dim3(NN), dim3(256), 0, stream>>>(act, pw1, pb1, pw2, pb2, bat, sN, mG);
  pool_exp<<<dim3((NN + 255) / 256), dim3(256), 0, stream>>>(sN, bat, mG, exN, denG);
  pool_weighted<<<dim3((NN * HIDDEN + 255) / 256), dim3(256), 0, stream>>>(
      act, bat, exN, denG, ge);
  classify<<<dim3(NG), dim3(256), 0, stream>>>(ge, cw1, cb1, cw2, cb2, out);
}
